// MetaBertSelfAttention_25563645345864
// MI455X (gfx1250) — matrix operand order repaired and checked by reading, not compiled, not run on hardware
//
#include <hip/hip_runtime.h>
#include <hip/hip_bf16.h>

// Problem constants (match reference)
#define B_   8
#define S_   1024
#define H_   1024
#define NH_  16
#define D_   64
#define MTOT (B_ * S_)   // 8192 rows of hidden_states

typedef __bf16 bf16_t;
typedef __attribute__((ext_vector_type(16))) __bf16 v16bf;
typedef __attribute__((ext_vector_type(8)))  __bf16 v8bf;
typedef __attribute__((ext_vector_type(4)))  __bf16 v4bf;
typedef __attribute__((ext_vector_type(8)))  float  v8f;

// ---------------------------------------------------------------------------
// WMMA helper: D = A(16x32 bf16) * B(32x16 bf16) + C(16x16 f32)
// ---------------------------------------------------------------------------
__device__ __forceinline__ v8f wmma_bf16(v16bf a, v16bf b, v8f c) {
    return __builtin_amdgcn_wmma_f32_16x16x32_bf16(
        /*neg_a=*/false, a, /*neg_b=*/false, b,
        /*c_mod=*/(short)0, c, /*reuse_a=*/false, /*reuse_b=*/false);
}

// A-matrix 16x32 bf16 layout: lane m (0..15) = row M=m.
// lanes 0-15 hold K=kbase+0..15, lanes 16-31 hold K=kbase+16..31 (contiguous).
__device__ __forceinline__ v16bf frag_a_bf16(const bf16_t* __restrict__ row,
                                             int kbase, int hi8) {
    v16bf f = *(const v16bf*)(row + kbase + 2 * hi8);
    return f;
}

// B-matrix 32x16 bf16 layout: lane n (0..15) = col N=n.
// lanes 0-15 hold K=kbase+0..15, lanes 16-31 hold K=kbase+16..31 (contiguous).
__device__ __forceinline__ v16bf frag_b_bf16(const bf16_t* __restrict__ row,
                                             int kbase, int hi16) {
    return *(const v16bf*)(row + kbase + hi16);
}

// ---------------------------------------------------------------------------
// Async global->LDS staging (CDNA5, ASYNCcnt-tracked).
// LDS byte address comes from the vdst VGPR; each lane moves 16 bytes.
// ---------------------------------------------------------------------------
__device__ __forceinline__ void async_load_b128(unsigned lds_off, const void* gptr) {
    asm volatile("global_load_async_to_lds_b128 %0, %1, off"
                 :: "v"(lds_off), "v"((unsigned long long)(uintptr_t)gptr)
                 : "memory");
}
__device__ __forceinline__ void wait_async0() {
    asm volatile("s_wait_asynccnt 0x0" ::: "memory");
}
// Async loads complete in order: with 8 newer (next-block) ops outstanding,
// waiting to <=8 guarantees the current block's 8 ops have landed.
__device__ __forceinline__ void wait_async_le8() {
    asm volatile("s_wait_asynccnt 0x8" ::: "memory");
}

// ---------------------------------------------------------------------------
// Kernel 0: fp32 -> bf16 pack (prepass; amortized over many re-reads).
// ---------------------------------------------------------------------------
__global__ __launch_bounds__(256) void pack_bf16_kernel(
    const float* __restrict__ src, bf16_t* __restrict__ dst, int n4)
{
    const int i = blockIdx.x * blockDim.x + threadIdx.x;
    if (i < n4) {
        float4 v = ((const float4*)src)[i];
        v4bf o;
        o[0] = (bf16_t)v.x; o[1] = (bf16_t)v.y;
        o[2] = (bf16_t)v.z; o[3] = (bf16_t)v.w;
        ((v4bf*)dst)[i] = o;
    }
}

// ---------------------------------------------------------------------------
// Kernel 1: fused QKV projection, bf16 inputs.  Y = X @ W^T + b -> bf16.
// grid = (N/64, M/128, 3), block = 128 (4 waves); wave owns 32(M) x 64(N).
// Q, K stored row-major [M, H]; V stored TRANSPOSED as Vt[col, M] so the
// attention P*V B-operand becomes a contiguous LDS vector load.
// ---------------------------------------------------------------------------
__global__ __launch_bounds__(128) void qkv_gemm_kernel(
    const bf16_t* __restrict__ Xb, const bf16_t* __restrict__ Wb,
    const float* __restrict__ bq, const float* __restrict__ bk,
    const float* __restrict__ bv,
    bf16_t* __restrict__ Qb, bf16_t* __restrict__ Kb, bf16_t* __restrict__ Vt)
{
    const int lane    = threadIdx.x & 31;
    const int wave    = threadIdx.x >> 5;
    const int laneLow = lane & 15;
    const int hi8     = (lane >> 4) << 3;
    const int hi16    = (lane >> 4) << 4;

    const bf16_t* W = Wb + (size_t)blockIdx.z * H_ * H_;
    const float* bias = (blockIdx.z == 0) ? bq : (blockIdx.z == 1) ? bk : bv;

    const int m0 = blockIdx.y * 128 + wave * 32;
    const int n0 = blockIdx.x * 64;

    const bf16_t* arow0 = Xb + (size_t)(m0 + laneLow) * H_;
    const bf16_t* arow1 = arow0 + (size_t)16 * H_;
    const bf16_t* brow[4];
#pragma unroll
    for (int t = 0; t < 4; ++t)
        brow[t] = W + (size_t)(n0 + t * 16 + laneLow) * H_;

    v8f acc0[4] = {};
    v8f acc1[4] = {};

    for (int k = 0; k < H_; k += 32) {
        v16bf a0 = frag_a_bf16(arow0, k, hi8);
        v16bf a1 = frag_a_bf16(arow1, k, hi8);
#pragma unroll
        for (int t = 0; t < 4; ++t) {
            v16bf b = frag_b_bf16(brow[t], k, hi16);
            acc0[t] = wmma_bf16(a0, b, acc0[t]);
            acc1[t] = wmma_bf16(a1, b, acc1[t]);
        }
    }

    // C/D layout: vgpr r -> row m0 + r + hi8 ; laneLow -> column.
    if (blockIdx.z == 2) {
        // V: transposed store.  Each lane holds 8 consecutive rows of one
        // column -> one aligned 16B v8bf store per tile-half.
#pragma unroll
        for (int t = 0; t < 4; ++t) {
            const int col = n0 + t * 16 + laneLow;
            const float bb = bias[col];
            v8bf o0, o1;
#pragma unroll
            for (int r = 0; r < 8; ++r) {
                o0[r] = (bf16_t)(acc0[t][r] + bb);
                o1[r] = (bf16_t)(acc1[t][r] + bb);
            }
            bf16_t* base = Vt + (size_t)col * MTOT;
            *(v8bf*)(base + m0 + hi8)      = o0;
            *(v8bf*)(base + m0 + 16 + hi8) = o1;
        }
    } else {
        bf16_t* out = (blockIdx.z == 0) ? Qb : Kb;
#pragma unroll
        for (int t = 0; t < 4; ++t) {
            const int col = n0 + t * 16 + laneLow;
            const float bb = bias[col];
#pragma unroll
            for (int r = 0; r < 8; ++r) {
                out[(size_t)(m0 + r + hi8) * H_ + col]      = (bf16_t)(acc0[t][r] + bb);
                out[(size_t)(m0 + 16 + r + hi8) * H_ + col] = (bf16_t)(acc1[t][r] + bb);
            }
        }
    }
}

// ---------------------------------------------------------------------------
// Kernel 2: flash attention.  grid = (S/64, NH, B), block = 128 (4 waves).
// K blocks [64 kv x 64 d] and Vt blocks [64 d x 64 kv] double-buffered
// through LDS with global_load_async_to_lds_b128 (DMA overlapped with WMMA).
// ---------------------------------------------------------------------------
__global__ __launch_bounds__(128) void attn_kernel(
    const bf16_t* __restrict__ Qb, const bf16_t* __restrict__ Kb,
    const bf16_t* __restrict__ Vt, const float* __restrict__ mask,
    float* __restrict__ out)
{
    __shared__ bf16_t k_lds[2 * 64 * 64];    // K ping-pong (16KB)
    __shared__ bf16_t vt_lds[2 * 64 * 64];   // Vt ping-pong (16KB)
    __shared__ bf16_t p_lds[4 * 16 * 64];    // per-wave P tiles (8KB)

    const int lane    = threadIdx.x & 31;
    const int wave    = threadIdx.x >> 5;
    const int laneLow = lane & 15;
    const int hi8     = (lane >> 4) << 3;
    const int hi16    = (lane >> 4) << 4;

    const int b  = blockIdx.z;
    const int h  = blockIdx.y;
    const int q0 = blockIdx.x * 64 + wave * 16;

    const size_t headOff = (size_t)h * D_;
    const bf16_t* Qrow = Qb + ((size_t)(b * S_ + q0 + laneLow) * H_ + headOff);

    const unsigned k_base = (unsigned)(uintptr_t)&k_lds[0];
    const unsigned v_base = (unsigned)(uintptr_t)&vt_lds[0];

    // Issue async loads for the K and Vt blocks at kv offset j0 into `buf`.
    // 512 16B chunks per matrix; each thread issues 4 per matrix (8 total).
    auto stage = [&](int buf, int j0) {
        const char* Ksrc = (const char*)(Kb + ((size_t)(b * S_ + j0) * H_ + headOff));
        // Vt rows are head dims (h*64 + d), columns are kv positions.
        const char* Vsrc = (const char*)(Vt + ((size_t)(h * D_) * MTOT + (size_t)(b * S_ + j0)));
        const unsigned kb = k_base + (unsigned)(buf * 8192);
        const unsigned vb = v_base + (unsigned)(buf * 8192);
#pragma unroll
        for (int i = 0; i < 4; ++i) {
            const int c   = i * 128 + (int)threadIdx.x;  // chunk id 0..511
            const int row = c >> 3;
            const int bo  = (c & 7) * 16;                // byte offset in row
            const unsigned lo = (unsigned)(row * 128 + bo);
            async_load_b128(kb + lo, Ksrc + (size_t)row * (H_ * 2) + bo);
            async_load_b128(vb + lo, Vsrc + (size_t)row * (MTOT * 2) + bo);
        }
    };

    // Prologue: stage block 0 while we set up Q fragments.
    stage(0, 0);

    const v16bf aq0 = frag_a_bf16(Qrow, 0,  hi8);
    const v16bf aq1 = frag_a_bf16(Qrow, 32, hi8);

    v8f cacc[4] = {};
    float mrow[8], lrow[8];
#pragma unroll
    for (int r = 0; r < 8; ++r) { mrow[r] = -3.0e38f; lrow[r] = 0.0f; }

    bf16_t* pmine = p_lds + wave * (16 * 64);
    const float* maskb = mask + (size_t)b * S_;

    const int NBLK = S_ / 64;
    for (int jb = 0; jb < NBLK; ++jb) {
        const int j0  = jb * 64;
        const int buf = jb & 1;

        // Prefetch next block into the other buffer, then wait for current.
        if (jb + 1 < NBLK) { stage(buf ^ 1, j0 + 64); wait_async_le8(); }
        else               { wait_async0(); }
        __syncthreads();   // current K/Vt block visible to all waves

        const bf16_t* kbuf = k_lds + buf * 4096;
        const bf16_t* vbuf = vt_lds + buf * 4096;

        // Scores S = Q * K^T for 4 N-tiles of 16 kv rows each (K from LDS).
        float sc[4][8];
#pragma unroll
        for (int t = 0; t < 4; ++t) {
            const bf16_t* Krow = kbuf + (t * 16 + laneLow) * 64;
            v16bf bk0 = frag_b_bf16(Krow, 0,  hi16);
            v16bf bk1 = frag_b_bf16(Krow, 32, hi16);
            v8f st = {};
            st = wmma_bf16(aq0, bk0, st);
            st = wmma_bf16(aq1, bk1, st);
            const float mv = maskb[j0 + t * 16 + laneLow];
#pragma unroll
            for (int r = 0; r < 8; ++r)
                sc[t][r] = st[r] * 0.125f + mv;   // 1/sqrt(64) scale + mask
        }

        // Online softmax (row = vgpr r + hi8; row data across laneLow).
        float alpha[8];
#pragma unroll
        for (int r = 0; r < 8; ++r) {
            float mx = fmaxf(fmaxf(sc[0][r], sc[1][r]), fmaxf(sc[2][r], sc[3][r]));
            mx = fmaxf(mx, __shfl_xor(mx, 1, 32));
            mx = fmaxf(mx, __shfl_xor(mx, 2, 32));
            mx = fmaxf(mx, __shfl_xor(mx, 4, 32));
            mx = fmaxf(mx, __shfl_xor(mx, 8, 32));
            const float mnew = fmaxf(mrow[r], mx);
            alpha[r] = __expf(mrow[r] - mnew);
            mrow[r]  = mnew;
            float ls = 0.0f;
#pragma unroll
            for (int t = 0; t < 4; ++t) {
                const float p = __expf(sc[t][r] - mnew);
                sc[t][r] = p;
                ls += p;
            }
            ls += __shfl_xor(ls, 1, 32);
            ls += __shfl_xor(ls, 2, 32);
            ls += __shfl_xor(ls, 4, 32);
            ls += __shfl_xor(ls, 8, 32);
            lrow[r] = lrow[r] * alpha[r] + ls;
#pragma unroll
            for (int t2 = 0; t2 < 4; ++t2)
                cacc[t2][r] *= alpha[r];
        }

        // Stash P (bf16) into this wave's LDS tile.  Same-wave LDS RAW:
        // DS ops are in-order within a wave, no barrier needed before reload.
#pragma unroll
        for (int t = 0; t < 4; ++t)
#pragma unroll
            for (int r = 0; r < 8; ++r)
                pmine[(r + hi8) * 64 + t * 16 + laneLow] = (bf16_t)sc[t][r];

        // ctx += P (16x64) * V (64x64).  B-operand comes from Vt rows:
        // B[k][n] = V[j0+k][h*64+n] = Vt[h*64+n][b*S+j0+k] -> per-lane
        // kv-contiguous 32B vector load, same shape as the K fragments.
#pragma unroll
        for (int c = 0; c < 2; ++c) {
            v16bf ap = frag_a_bf16(pmine + laneLow * 64, c * 32, hi8);
#pragma unroll
            for (int t2 = 0; t2 < 4; ++t2) {
                v16bf bvf = frag_b_bf16(vbuf + (t2 * 16 + laneLow) * 64, c * 32, hi16);
                cacc[t2] = wmma_bf16(ap, bvf, cacc[t2]);
            }
        }

        __syncthreads();   // all waves done with `buf` before it is re-staged
    }

    // Epilogue: ctx / l -> out[b, q, h*64 + col] (fp32)
#pragma unroll
    for (int t2 = 0; t2 < 4; ++t2) {
        const int col = h * 64 + t2 * 16 + laneLow;
#pragma unroll
        for (int r = 0; r < 8; ++r) {
            const int q = q0 + r + hi8;
            out[(size_t)(b * S_ + q) * H_ + col] = cacc[t2][r] / lrow[r];
        }
    }
}

// ---------------------------------------------------------------------------
extern "C" void kernel_launch(void* const* d_in, const int* in_sizes, int n_in,
                              void* d_out, int out_size, void* d_ws, size_t ws_size,
                              hipStream_t stream) {
    const float* X    = (const float*)d_in[0];
    const float* mask = (const float*)d_in[1];
    const float* Wq   = (const float*)d_in[2];
    const float* bq   = (const float*)d_in[3];
    const float* Wk   = (const float*)d_in[4];
    const float* bk   = (const float*)d_in[5];
    const float* Wv   = (const float*)d_in[6];
    const float* bv   = (const float*)d_in[7];
    float* out = (float*)d_out;

    const size_t elems = (size_t)B_ * S_ * H_;   // 8M elements
    const size_t wsz   = (size_t)H_ * H_;        // 1M elements per weight

    bf16_t* Qb = (bf16_t*)d_ws;                  // 16 MB
    bf16_t* Kb = Qb + elems;                     // 16 MB
    bf16_t* Vt = Kb + elems;                     // 16 MB (transposed V)
    bf16_t* Xb = Vt + elems;                     // 16 MB
    bf16_t* Wb = Xb + elems;                     //  6 MB (Wq|Wk|Wv)

    // Pack fp32 -> bf16 (X and the three weight matrices).
    {
        const int n4x = (int)(elems / 4);
        const int n4w = (int)(wsz / 4);
        pack_bf16_kernel<<<(n4x + 255) / 256, 256, 0, stream>>>(X,  Xb,           n4x);
        pack_bf16_kernel<<<(n4w + 255) / 256, 256, 0, stream>>>(Wq, Wb,           n4w);
        pack_bf16_kernel<<<(n4w + 255) / 256, 256, 0, stream>>>(Wk, Wb + wsz,     n4w);
        pack_bf16_kernel<<<(n4w + 255) / 256, 256, 0, stream>>>(Wv, Wb + 2 * wsz, n4w);
    }

    // QKV projections: M=8192 (128-row blocks), N=1024 -> grid (16, 64, 3)
    qkv_gemm_kernel<<<dim3(H_ / 64, MTOT / 128, 3), 128, 0, stream>>>(
        Xb, Wb, bq, bk, bv, Qb, Kb, Vt);

    // Flash attention: grid (S/64, heads, batch)
    attn_kernel<<<dim3(S_ / 64, NH_, B_), 128, 0, stream>>>(
        Qb, Kb, Vt, mask, out);
}
